// GFP_55671366091116
// MI455X (gfx1250) — matrix operand order repaired and checked by reading, not compiled, not run on hardware
//
#include <hip/hip_runtime.h>
#include <math.h>

typedef __attribute__((ext_vector_type(16))) _Float16     v16h;
typedef __attribute__((ext_vector_type(4)))  _Float16     v4h;
typedef __attribute__((ext_vector_type(8)))  float        v8f;
typedef __attribute__((ext_vector_type(4)))  unsigned int v4u;
typedef __attribute__((ext_vector_type(8)))  int          v8i;
typedef __attribute__((ext_vector_type(4)))  int          v4i;

#define B_   4
#define N_   4096
#define C_   128
#define EPSL 1e-6f
#define TM   16                 // rows per workgroup (one WMMA M-tile)
#define CK   64                 // j-chunk = two K=32 WMMA sub-steps
#define NCHUNK (N_ / CK)
#define NBUF 4                  // TDM pipeline depth (issue 3 ahead)

// ---------------------------------------------------------------------------
// TDM: 2D tile load adj[i0:i0+TM, j0:j0+CK] (f32, row stride N_) -> LDS.
// D# per CDNA5 ISA 8.3/8.4: group0 {count=1, lds_addr, global_addr, type=2},
// group1 {data_size=4B, tensor_dim0/1=N_, tile 64x16, dim0_stride=N_}.
// All inputs wave-uniform -> descriptor lives in SGPRs.
// ---------------------------------------------------------------------------
__device__ __forceinline__ void tdm_load_tile(const float* gptr, unsigned lds_off)
{
    unsigned long long ga = (unsigned long long)(uintptr_t)gptr;
    v4u g0;
    g0[0] = 1u;                                           // count=1 (valid), user mode
    g0[1] = lds_off;                                      // LDS byte address
    g0[2] = (unsigned)(ga & 0xFFFFFFFFu);                 // global_addr[31:0]
    g0[3] = (unsigned)((ga >> 32) & 0x01FFFFFFu)          // global_addr[56:32]
          | (2u << 30);                                   // type=2 ("image")
    v8i g1;
    g1[0] = (int)(2u << 16);                              // data_size=2 (4B); mask/flags=0
    g1[1] = (int)((unsigned)(N_ & 0xFFFF) << 16);         // tensor_dim0[15:0]
    g1[2] = (int)((unsigned)(N_ >> 16)
          | ((unsigned)(N_ & 0xFFFF) << 16));             // dim0[31:16] | dim1[15:0]
    g1[3] = (int)((unsigned)(N_ >> 16)
          | ((unsigned)CK << 16));                        // dim1[31:16] | tile_dim0=64
    g1[4] = (int)TM;                                      // tile_dim1=16 | tile_dim2=0
    g1[5] = (int)N_;                                      // tensor_dim0_stride low32
    g1[6] = 0;                                            // stride0 hi | stride1 lo
    g1[7] = 0;                                            // stride1 hi
    v4i gz = {0, 0, 0, 0};                                // 2D: groups 2/3 unused
#if __clang_major__ >= 23
    v8i gz8 = {0, 0, 0, 0, 0, 0, 0, 0};
    __builtin_amdgcn_tensor_load_to_lds(g0, g1, gz, gz, gz8, 0);
#else
    __builtin_amdgcn_tensor_load_to_lds(g0, g1, gz, gz, 0);
#endif
}

// ---------------------------------------------------------------------------
// Kernel 1: per-row s2 = x . a2 (s1 cancels inside the row softmax), plus
// fp16 copy of x for the P*V WMMA. One wave per row.
// ---------------------------------------------------------------------------
__global__ __launch_bounds__(256)
void gat_pre(const float* __restrict__ x, const float* __restrict__ a,
             float* __restrict__ s2, _Float16* __restrict__ xh)
{
    const int lane = threadIdx.x & 31;
    const int wv   = threadIdx.x >> 5;
    const int row  = blockIdx.x * 8 + wv;            // flat (b*N + n)

    const float4 xv = *reinterpret_cast<const float4*>(x + (size_t)row * C_ + lane * 4);
    const float4 a2 = *reinterpret_cast<const float4*>(a + C_ + lane * 4);

    float d2 = xv.x * a2.x + xv.y * a2.y + xv.z * a2.z + xv.w * a2.w;
    #pragma unroll
    for (int m = 16; m > 0; m >>= 1)
        d2 += __shfl_xor(d2, m, 32);
    if (lane == 0) s2[row] = d2;

    v4h h;
    h[0] = (_Float16)xv.x; h[1] = (_Float16)xv.y;
    h[2] = (_Float16)xv.z; h[3] = (_Float16)xv.w;
    *reinterpret_cast<v4h*>(xh + (size_t)row * C_ + lane * 4) = h;
}

// ---------------------------------------------------------------------------
// Kernel 2: fused flash-style attention + residual + LayerNorm.
// 128 threads = 4 waves; wave w owns output columns [32w, 32w+32).
// Wave 0: TDM pipeline (3 tiles in flight) + softmax producer.
// Waves 0-3: consume shared A-fragment, rescale, 4x WMMA per chunk.
// ---------------------------------------------------------------------------
__global__ __launch_bounds__(128)
void gat_main(const float* __restrict__ x,  const float* __restrict__ adj,
              const float* __restrict__ s2g,
              const _Float16* __restrict__ xh,
              const float* __restrict__ gamma, const float* __restrict__ beta,
              float* __restrict__ out)
{
    __shared__ alignas(16) float    lds_adj[NBUF][TM][CK];  // TDM destinations
    __shared__ alignas(16) float    lds_s2[N_ == 4096 ? 4096 : N_]; // whole batch slice
    __shared__ alignas(32) _Float16 afrag_lds[2][32][16];   // A fragments, 2 sub-steps
    __shared__ alignas(16) float    alpha_lds[16];          // per-row rescale
    __shared__ alignas(16) float    inv_lds[16];            // per-row 1/lsum
    __shared__ alignas(16) float    ytile[TM][C_];          // epilogue staging

    const int t    = threadIdx.x;
    const int lane = t & 31;
    const int wv   = t >> 5;
    const int b    = blockIdx.x / (N_ / TM);
    const int i0   = (blockIdx.x % (N_ / TM)) * TM;

    // A-fragment lane mapping (16-bit A 16x32, ISA 7.12.2):
    //   lane<16: M=lane,    K in {0..7} then {16..23}
    //   lane>=16:M=lane-16, K in {8..15} then {24..31}
    const int r   = lane & 15;
    const int hi  = lane >> 4;
    const int kb0 = hi * 8;
    const int kb1 = 16 + hi * 8;
    const int c0  = wv * 32;               // this wave's column base

    const float* adj_row = adj + (size_t)b * N_ * N_ + (size_t)i0 * N_;
    const float* s2_base = s2g + b * N_;
    const unsigned lds_adj_off = (unsigned)(uintptr_t)&lds_adj[0][0][0];

    v8f acc0 = {}; v8f acc1 = {};
    float mrow = -1e30f, lsum = 0.0f;

    // --- prologue: stage s2[b] into LDS; launch TDM pipeline ---------------
    {
        const int base = t * (N_ / 128);   // 32 floats per thread
        #pragma unroll
        for (int q = 0; q < (N_ / 128) / 4; ++q) {
            float4 v = *reinterpret_cast<const float4*>(s2_base + base + 4 * q);
            *reinterpret_cast<float4*>(&lds_s2[base + 4 * q]) = v;
        }
    }
    if (wv == 0) {
        #pragma unroll
        for (int k = 0; k < NBUF - 1; ++k)   // issue chunks 0..2
            tdm_load_tile(adj_row + k * CK,
                          lds_adj_off + (unsigned)k * (TM * CK * 4));
    }
    __syncthreads();   // lds_s2 ready

    // --- main j-stream ------------------------------------------------------
    for (int ch = 0; ch < NCHUNK; ++ch) {
        const int j0 = ch * CK;

        // B fragments first: in flight while softmax runs / waves wait.
        // 32x16 f16 B layout: lane = N column (c0 [+16] + lane%16), K in same order as A.
        const _Float16* xcol = xh + ((size_t)b * N_ + j0) * C_ + c0 + r;
        v16h bf00, bf01, bf10, bf11;
        #pragma unroll
        for (int e = 0; e < 16; ++e) {
            const int kk = (e < 8) ? (kb0 + e) : (kb1 + e - 8);
            const _Float16* pk = xcol + (size_t)kk * C_;
            bf00[e] = pk[0];
            bf01[e] = pk[16];
            bf10[e] = pk[(size_t)32 * C_];
            bf11[e] = pk[(size_t)32 * C_ + 16];
        }

        // ---- wave 0: keep TDM pipeline full, then online softmax ----------
        if (wv == 0) {
            if (ch + NBUF - 1 < NCHUNK)
                tdm_load_tile(adj_row + (ch + NBUF - 1) * CK,
                              lds_adj_off + (unsigned)((ch + NBUF - 1) & (NBUF - 1)) * (TM * CK * 4));
            // wait for chunk ch (TDM completes in order within this wave)
            const int rem = NCHUNK - 1 - ch;
            if      (rem >= 3) __builtin_amdgcn_s_wait_tensorcnt(3);
            else if (rem == 2) __builtin_amdgcn_s_wait_tensorcnt(2);
            else if (rem == 1) __builtin_amdgcn_s_wait_tensorcnt(1);
            else               __builtin_amdgcn_s_wait_tensorcnt(0);

            const int cur = ch & (NBUF - 1);
            float p[32];
            float cmax = -1e30f;
            #pragma unroll
            for (int s = 0; s < 2; ++s) {
                const float* tb = &lds_adj[cur][r][s * 32];
                const float* sb = &lds_s2[j0 + s * 32];
                #pragma unroll
                for (int e = 0; e < 8; ++e) {
                    float sc = tb[kb0 + e] + sb[kb0 + e];
                    p[s*16 + e] = sc; cmax = fmaxf(cmax, sc);
                }
                #pragma unroll
                for (int e = 0; e < 8; ++e) {
                    float sc = tb[kb1 + e] + sb[kb1 + e];
                    p[s*16 + 8 + e] = sc; cmax = fmaxf(cmax, sc);
                }
            }
            cmax = fmaxf(cmax, __shfl_xor(cmax, 16, 32));   // row r lives in lanes r, r+16
            const float mnew  = fmaxf(mrow, cmax);
            const float alpha = __expf(mrow - mnew);

            float psum = 0.f;
            v16h af0, af1;
            #pragma unroll
            for (int e = 0; e < 16; ++e) {
                float pe = __expf(p[e] - mnew);
                psum += pe; af0[e] = (_Float16)pe;
            }
            #pragma unroll
            for (int e = 0; e < 16; ++e) {
                float pe = __expf(p[16 + e] - mnew);
                psum += pe; af1[e] = (_Float16)pe;
            }
            psum += __shfl_xor(psum, 16, 32);
            lsum  = lsum * alpha + psum;
            mrow  = mnew;

            *reinterpret_cast<v16h*>(&afrag_lds[0][lane][0]) = af0;
            *reinterpret_cast<v16h*>(&afrag_lds[1][lane][0]) = af1;
            if (lane < 16) alpha_lds[lane] = alpha;
        }
        __syncthreads();   // A fragments + alpha published

        // ---- all waves: rescale accumulators, consume A, do 4 WMMAs -------
        {
            float4 av0 = *reinterpret_cast<const float4*>(&alpha_lds[8 * hi]);
            float4 av1 = *reinterpret_cast<const float4*>(&alpha_lds[8 * hi + 4]);
            const float a8[8] = {av0.x, av0.y, av0.z, av0.w, av1.x, av1.y, av1.z, av1.w};
            #pragma unroll
            for (int g = 0; g < 8; ++g) { acc0[g] *= a8[g]; acc1[g] *= a8[g]; }
        }
        v16h af0 = *reinterpret_cast<const v16h*>(&afrag_lds[0][lane][0]);
        v16h af1 = *reinterpret_cast<const v16h*>(&afrag_lds[1][lane][0]);

        acc0 = __builtin_amdgcn_wmma_f32_16x16x32_f16(false, af0, false, bf00,
                                                      (short)0, acc0, false, false);
        acc1 = __builtin_amdgcn_wmma_f32_16x16x32_f16(false, af0, false, bf01,
                                                      (short)0, acc1, false, false);
        acc0 = __builtin_amdgcn_wmma_f32_16x16x32_f16(false, af1, false, bf10,
                                                      (short)0, acc0, false, false);
        acc1 = __builtin_amdgcn_wmma_f32_16x16x32_f16(false, af1, false, bf11,
                                                      (short)0, acc1, false, false);

        __syncthreads();   // A fragments consumed; producer may overwrite
    }

    // --- epilogue: publish 1/lsum, normalize, stage y-tile in LDS -----------
    if (wv == 0 && lane < 16) inv_lds[lane] = 1.0f / lsum;
    __syncthreads();
    {
        float4 iv0 = *reinterpret_cast<const float4*>(&inv_lds[8 * hi]);
        float4 iv1 = *reinterpret_cast<const float4*>(&inv_lds[8 * hi + 4]);
        const float i8[8] = {iv0.x, iv0.y, iv0.z, iv0.w, iv1.x, iv1.y, iv1.z, iv1.w};
        #pragma unroll
        for (int g = 0; g < 8; ++g) {
            ytile[g + 8 * hi][c0 +      (lane & 15)] = acc0[g] * i8[g];
            ytile[g + 8 * hi][c0 + 16 + (lane & 15)] = acc1[g] * i8[g];
        }
    }
    __syncthreads();

    // --- residual + LayerNorm: 8 threads per row, 16 cols each --------------
    const int row  = t >> 3;
    const int cseg = (t & 7) * 16;
    const float* xr = x + ((size_t)b * N_ + i0 + row) * C_ + cseg;

    float yv[16];
    float s = 0.f, sq = 0.f;
    #pragma unroll
    for (int q = 0; q < 4; ++q) {
        float4 yq = *reinterpret_cast<const float4*>(&ytile[row][cseg + 4 * q]);
        float4 xq = *reinterpret_cast<const float4*>(xr + 4 * q);
        float v0 = yq.x + xq.x, v1 = yq.y + xq.y, v2 = yq.z + xq.z, v3 = yq.w + xq.w;
        yv[4*q+0] = v0; yv[4*q+1] = v1; yv[4*q+2] = v2; yv[4*q+3] = v3;
        s  += v0 + v1 + v2 + v3;
        sq += v0 * v0 + v1 * v1 + v2 * v2 + v3 * v3;
    }
    #pragma unroll
    for (int m = 4; m > 0; m >>= 1) {   // reduce across the 8 lanes of this row
        s  += __shfl_xor(s,  m, 32);
        sq += __shfl_xor(sq, m, 32);
    }
    const float mu   = s * (1.0f / C_);
    const float var  = sq * (1.0f / C_) - mu * mu;
    const float rstd = rsqrtf(var + EPSL);

    float* orow = out + ((size_t)b * N_ + i0 + row) * C_ + cseg;
    #pragma unroll
    for (int q = 0; q < 4; ++q) {
        float4 gq = *reinterpret_cast<const float4*>(gamma + cseg + 4 * q);
        float4 bq = *reinterpret_cast<const float4*>(beta  + cseg + 4 * q);
        float4 o;
        o.x = (yv[4*q+0] - mu) * rstd * gq.x + bq.x;
        o.y = (yv[4*q+1] - mu) * rstd * gq.y + bq.y;
        o.z = (yv[4*q+2] - mu) * rstd * gq.z + bq.z;
        o.w = (yv[4*q+3] - mu) * rstd * gq.w + bq.w;
        *reinterpret_cast<float4*>(orow + 4 * q) = o;
    }
}

// ---------------------------------------------------------------------------
extern "C" void kernel_launch(void* const* d_in, const int* in_sizes, int n_in,
                              void* d_out, int out_size, void* d_ws, size_t ws_size,
                              hipStream_t stream)
{
    const float* x     = (const float*)d_in[0];   // (B,N,C)
    const float* adj   = (const float*)d_in[1];   // (B,N,N)
    const float* a     = (const float*)d_in[2];   // (2C,)
    const float* gamma = (const float*)d_in[3];   // (C,)
    const float* beta  = (const float*)d_in[4];   // (C,)
    float*       out   = (float*)d_out;

    // workspace layout: s2 (B*N f32) | xh (B*N*C f16) ~= 4.06 MB
    float*    s2 = (float*)d_ws;
    _Float16* xh = (_Float16*)(s2 + (size_t)B_ * N_);

    gat_pre <<<B_ * N_ / 8,    256, 0, stream>>>(x, a, s2, xh);
    gat_main<<<B_ * (N_ / TM), 128, 0, stream>>>(x, adj, s2, xh, gamma, beta, out);
}
